// MultiHeadAttention_52750788329621
// MI455X (gfx1250) — matrix operand order repaired and checked by reading, not compiled, not run on hardware
//
#include <hip/hip_runtime.h>
#include <hip/hip_bf16.h>
#include <math.h>

typedef __attribute__((ext_vector_type(16))) _Float16 v16h;
typedef __attribute__((ext_vector_type(8)))  _Float16 v8h;
typedef __attribute__((ext_vector_type(4)))  _Float16 v4h;
typedef __attribute__((ext_vector_type(8)))  float    v8f;
typedef __attribute__((ext_vector_type(4)))  int      v4i;

#define D_MODEL 1024
#define S_LEN   2048
#define BATCH   2
#define NHEAD   16
#define DK      64
#define M_ROWS  (BATCH * S_LEN)   // 4096

#if defined(__has_builtin)
#if __has_builtin(__builtin_amdgcn_global_load_async_to_lds_b128) && \
    __has_builtin(__builtin_amdgcn_s_wait_asynccnt)
#define USE_ASYNC_LDS 1
#endif
#endif
#ifndef USE_ASYNC_LDS
#define USE_ASYNC_LDS 0
#endif

#if USE_ASYNC_LDS
typedef __attribute__((address_space(1))) v4i* g_v4i_p;
typedef __attribute__((address_space(3))) v4i* l_v4i_p;
#endif

static __device__ __forceinline__ v8f wmma_f16(v16h a, v16h b, v8f c) {
    // D = A(16x32 f16) * B(32x16 f16) + C(16x16 f32)
    return __builtin_amdgcn_wmma_f32_16x16x32_f16(
        /*neg_a=*/false, a, /*neg_b=*/false, b,
        /*c_mod=*/(short)0, c, /*reuse_a=*/false, /*reuse_b=*/false);
}

static __device__ __forceinline__ v16h concat8(v8h lo, v8h hi8) {
    return __builtin_shufflevector(lo, hi8, 0,1,2,3,4,5,6,7,8,9,10,11,12,13,14,15);
}

// -------------------------------------------------------------------------
// f32 -> f16 bulk convert, 8 elements / thread (each element converted ONCE)
// -------------------------------------------------------------------------
__global__ __launch_bounds__(256) void cvt_f16_kernel(
    const float* __restrict__ in, _Float16* __restrict__ out)
{
    size_t base = ((size_t)blockIdx.x * blockDim.x + threadIdx.x) * 8;
    float4 f0 = *(const float4*)(in + base);
    float4 f1 = *(const float4*)(in + base + 4);
    v8h o;
    o[0] = (_Float16)f0.x; o[1] = (_Float16)f0.y; o[2] = (_Float16)f0.z; o[3] = (_Float16)f0.w;
    o[4] = (_Float16)f1.x; o[5] = (_Float16)f1.y; o[6] = (_Float16)f1.z; o[7] = (_Float16)f1.w;
    *(v8h*)(out + base) = o;
}

// -------------------------------------------------------------------------
// GEMM (all-f16 operands): Y[4096,1024] = Xh[4096,1024] @ Wh[1024,1024]^T
// Block = 8 waves sharing one 64-wide N block; each wave owns 32 M rows.
// B tile (64n x 32k f16 = 4KB) double-buffered in LDS via CDNA5 async
// global->LDS DMA (ASYNCcnt); next-tile DMA overlaps a full k-step of WMMA.
// mode 0: f32 Y[m,n].  mode 1: f16 transposed Yt[(b*1024+n)*2048+s]  (for V)
// -------------------------------------------------------------------------
#define BPAD 40   // LDS row pitch in halfs (80B): conflict-free for r=0..15

__global__ __launch_bounds__(256) void gemm_f16_kernel(
    const _Float16* __restrict__ Xh, const _Float16* __restrict__ Wh,
    float* __restrict__ Yf, _Float16* __restrict__ Yt, int mode)
{
    const int lane = threadIdx.x & 31;
    const int r    = lane & 15;
    const int hi   = lane >> 4;
    const int w    = threadIdx.x >> 5;
    const int m0   = (((blockIdx.x >> 4) << 3) + w) << 5;   // 128 m-tiles of 32
    const int n0   = (blockIdx.x & 15) << 6;                // 16 n-blocks of 64 (shared)

    v8f acc[2][4] = {{v8f{}, v8f{}, v8f{}, v8f{}}, {v8f{}, v8f{}, v8f{}, v8f{}}};
    const _Float16* x0 = Xh + (size_t)(m0 + r) * D_MODEL;
    const _Float16* x1 = Xh + (size_t)(m0 + 16 + r) * D_MODEL;

#if USE_ASYNC_LDS
    __shared__ _Float16 bsh[2][64 * BPAD];                  // 2 x 5KB
    const int srow = threadIdx.x >> 2;                      // 0..63
    const int sseg = threadIdx.x & 3;                       // 0..3 (16B each)
    const _Float16* gsrc = Wh + (size_t)(n0 + srow) * D_MODEL + sseg * 8;

    __builtin_amdgcn_global_load_async_to_lds_b128(
        (g_v4i_p)(void*)gsrc,
        (l_v4i_p)&bsh[0][srow * BPAD + sseg * 8], 0, 0);

    for (int ks = 0; ks < D_MODEL / 32; ++ks) {
        const int kk  = ks << 5;
        const int cur = ks & 1;
        __builtin_amdgcn_s_wait_asynccnt(0);   // own DMA for 'cur' complete
        __syncthreads();                       // all waves' DMA done & prior reads done
        if (ks + 1 < D_MODEL / 32) {
            __builtin_amdgcn_global_load_async_to_lds_b128(
                (g_v4i_p)(void*)(gsrc + kk + 32),
                (l_v4i_p)&bsh[cur ^ 1][srow * BPAD + sseg * 8], 0, 0);
        }
        __builtin_prefetch(x0 + kk + 128, 0, 1);
        __builtin_prefetch(x1 + kk + 128, 0, 1);
        v16h a0 = concat8(*(const v8h*)(x0 + kk + hi * 8),
                          *(const v8h*)(x0 + kk + 16 + hi * 8));
        v16h a1 = concat8(*(const v8h*)(x1 + kk + hi * 8),
                          *(const v8h*)(x1 + kk + 16 + hi * 8));
#pragma unroll
        for (int nt = 0; nt < 4; ++nt) {
            const _Float16* bp = &bsh[cur][(nt * 16 + r) * BPAD + hi * 8];
            v16h b = concat8(*(const v8h*)bp, *(const v8h*)(bp + 16));
            acc[0][nt] = wmma_f16(a0, b, acc[0][nt]);
            acc[1][nt] = wmma_f16(a1, b, acc[1][nt]);
        }
    }
#else
    for (int kk = 0; kk < D_MODEL; kk += 32) {
        __builtin_prefetch(x0 + kk + 128, 0, 1);
        __builtin_prefetch(x1 + kk + 128, 0, 1);
        v16h a0 = concat8(*(const v8h*)(x0 + kk + hi * 8),
                          *(const v8h*)(x0 + kk + 16 + hi * 8));
        v16h a1 = concat8(*(const v8h*)(x1 + kk + hi * 8),
                          *(const v8h*)(x1 + kk + 16 + hi * 8));
#pragma unroll
        for (int nt = 0; nt < 4; ++nt) {
            v16h b = concat8(*(const v8h*)(Wh + (size_t)(n0 + nt * 16 + r) * D_MODEL + kk + hi * 8),
                             *(const v8h*)(Wh + (size_t)(n0 + nt * 16 + r) * D_MODEL + kk + 16 + hi * 8));
            acc[0][nt] = wmma_f16(a0, b, acc[0][nt]);
            acc[1][nt] = wmma_f16(a1, b, acc[1][nt]);
        }
    }
#endif

    if (mode == 0) {
#pragma unroll
        for (int mt = 0; mt < 2; ++mt)
#pragma unroll
            for (int nt = 0; nt < 4; ++nt)
#pragma unroll
                for (int v = 0; v < 8; ++v)
                    Yf[(size_t)(m0 + mt * 16 + v + 8 * hi) * D_MODEL + n0 + nt * 16 + r]
                        = acc[mt][nt][v];
    } else {
        // transposed f16 store for V: Yt[(b*1024 + d)*2048 + s]
#pragma unroll
        for (int mt = 0; mt < 2; ++mt)
#pragma unroll
            for (int nt = 0; nt < 4; ++nt)
#pragma unroll
                for (int v = 0; v < 8; ++v) {
                    int m = m0 + mt * 16 + v + 8 * hi;
                    int bb = m >> 11, s = m & 2047;
                    int d = n0 + nt * 16 + r;
                    Yt[((size_t)(bb * D_MODEL + d) << 11) + s] = (_Float16)acc[mt][nt][v];
                }
    }
}

// -------------------------------------------------------------------------
// LayerNorm over d_model=1024, wave per row, output f16 in [b,h,s,dk] layout
// -------------------------------------------------------------------------
__global__ __launch_bounds__(256) void ln_to_f16_kernel(
    const float* __restrict__ Xf, const float* __restrict__ gamma,
    const float* __restrict__ beta, _Float16* __restrict__ Out)
{
    const int lane = threadIdx.x & 31;
    const int row  = blockIdx.x * 8 + (threadIdx.x >> 5);   // 4096 rows
    const float* xp = Xf + (size_t)row * D_MODEL;

    float4 xv[8];
    float s = 0.f, ss = 0.f;
#pragma unroll
    for (int i = 0; i < 8; ++i) {
        xv[i] = ((const float4*)xp)[i * 32 + lane];
        s  += xv[i].x + xv[i].y + xv[i].z + xv[i].w;
        ss += xv[i].x * xv[i].x + xv[i].y * xv[i].y + xv[i].z * xv[i].z + xv[i].w * xv[i].w;
    }
#pragma unroll
    for (int off = 16; off > 0; off >>= 1) {
        s  += __shfl_xor(s,  off, 32);
        ss += __shfl_xor(ss, off, 32);
    }
    const float mean = s * (1.0f / D_MODEL);
    const float var  = ss * (1.0f / D_MODEL) - mean * mean;
    const float rs   = rsqrtf(var + 1e-5f);
    const int b = row >> 11, sq = row & 2047;

#pragma unroll
    for (int i = 0; i < 8; ++i) {
        int e = i * 128 + lane * 4;
        float4 g  = ((const float4*)gamma)[e >> 2];
        float4 bt = ((const float4*)beta)[e >> 2];
        int h = e >> 6, dk = e & 63;
        v4h o;
        o[0] = (_Float16)((xv[i].x - mean) * rs * g.x + bt.x);
        o[1] = (_Float16)((xv[i].y - mean) * rs * g.y + bt.y);
        o[2] = (_Float16)((xv[i].z - mean) * rs * g.z + bt.z);
        o[3] = (_Float16)((xv[i].w - mean) * rs * g.w + bt.w);
        *(v4h*)(Out + ((size_t)(b * NHEAD + h) * S_LEN + sq) * DK + dk) = o;
    }
}

// -------------------------------------------------------------------------
// Flash attention: block = 8 waves, ALL on the same (b,h); each wave owns 16
// queries. K chunk (32x64) and V chunk (64x32) are shared by all 8 waves ->
// double-buffered in LDS via async global->LDS DMA (2 ops/thread/chunk),
// cutting K/V global traffic 8x. scores^T = K@Q^T so the C-layout IS the
// A-operand layout of P for P@V (no cross-lane transpose). Online softmax
// in-lane (+1 xor16 shuffle); ctx f32 accum; f16 ctx output.
// -------------------------------------------------------------------------
#define KPITCH 72   // 144B row pitch: conflict-free 16B reads across r
#define VPITCH 40   // 80B row pitch

__global__ __launch_bounds__(256) void attn_kernel(
    const _Float16* __restrict__ Qh, const _Float16* __restrict__ Kh,
    const _Float16* __restrict__ Vt, _Float16* __restrict__ Ch)
{
    const int lane = threadIdx.x & 31;
    const int r    = lane & 15;
    const int hi   = lane >> 4;
    const int wid  = blockIdx.x * 8 + (threadIdx.x >> 5);   // 4096 waves
    const int bh   = wid >> 7;                              // same for whole block
    const int q0   = (wid & 127) << 4;
    const int b    = bh >> 4, h = bh & 15;

    // Q^T B-operand: lane holds query q0+r, elem e: d = dstep*32 + hi*8 + e (e<8), dstep*32 + 16 + hi*8 + (e-8) (e>=8)
    const _Float16* qbase = Qh + ((size_t)bh * S_LEN + q0 + r) * DK;
    const v16h bq0 = concat8(*(const v8h*)(qbase + hi * 8), *(const v8h*)(qbase + 16 + hi * 8));
    const v16h bq1 = concat8(*(const v8h*)(qbase + 32 + hi * 8), *(const v8h*)(qbase + 48 + hi * 8));

    v8f ctx[4] = {v8f{}, v8f{}, v8f{}, v8f{}};
    float m = -__builtin_inff(), l = 0.f;
    const float cs = 0.125f * 1.44269504088896340736f;  // 1/sqrt(64) * log2(e)

#if USE_ASYNC_LDS
    __shared__ _Float16 ksh[2][32 * KPITCH];   // 2 x 4.5KB
    __shared__ _Float16 vsh[2][64 * VPITCH];   // 2 x 5KB
    const int t    = threadIdx.x;
    const int krow = t >> 3, kseg = t & 7;     // 32 rows x 8 x16B segs
    const int vrow = t >> 2, vseg = t & 3;     // 64 rows x 4 x16B segs
    const _Float16* kdma = Kh + ((size_t)bh * S_LEN + krow) * DK + kseg * 8;
    const _Float16* vdma = Vt + (size_t)(b * D_MODEL + h * DK + vrow) * S_LEN + vseg * 8;

    __builtin_amdgcn_global_load_async_to_lds_b128(
        (g_v4i_p)(void*)kdma, (l_v4i_p)&ksh[0][krow * KPITCH + kseg * 8], 0, 0);
    __builtin_amdgcn_global_load_async_to_lds_b128(
        (g_v4i_p)(void*)vdma, (l_v4i_p)&vsh[0][vrow * VPITCH + vseg * 8], 0, 0);

    for (int ic = 0; ic < S_LEN / 32; ++ic) {
        const int kc  = ic << 5;
        const int cur = ic & 1;
        __builtin_amdgcn_s_wait_asynccnt(0);
        __syncthreads();
        if (ic + 1 < S_LEN / 32) {
            __builtin_amdgcn_global_load_async_to_lds_b128(
                (g_v4i_p)(void*)(kdma + (size_t)(kc + 32) * DK),
                (l_v4i_p)&ksh[cur ^ 1][krow * KPITCH + kseg * 8], 0, 0);
            __builtin_amdgcn_global_load_async_to_lds_b128(
                (g_v4i_p)(void*)(vdma + kc + 32),
                (l_v4i_p)&vsh[cur ^ 1][vrow * VPITCH + vseg * 8], 0, 0);
        }
        v8f st[2];
#pragma unroll
        for (int tt = 0; tt < 2; ++tt) {
            const _Float16* kb = &ksh[cur][(tt * 16 + r) * KPITCH];
            v16h a0 = concat8(*(const v8h*)(kb + hi * 8),
                              *(const v8h*)(kb + 16 + hi * 8));
            v16h a1 = concat8(*(const v8h*)(kb + 32 + hi * 8),
                              *(const v8h*)(kb + 48 + hi * 8));
            v8f z = {};
            z = wmma_f16(a0, bq0, z);
            z = wmma_f16(a1, bq1, z);
            st[tt] = z;
        }
#else
    const _Float16* kbase  = Kh + ((size_t)bh * S_LEN + r) * DK;
    const _Float16* vbase0 = Vt + (size_t)(b * D_MODEL + h * DK) * S_LEN + hi * 8;
    for (int ic = 0; ic < S_LEN / 32; ++ic) {
        const int kc = ic << 5;
        __builtin_prefetch(kbase + (size_t)(kc + 32) * DK, 0, 1);
        __builtin_prefetch(vbase0 + (size_t)r * S_LEN + kc + 32, 0, 1);
        v8f st[2];
#pragma unroll
        for (int tt = 0; tt < 2; ++tt) {
            const _Float16* kb = kbase + (size_t)(kc + tt * 16) * DK;
            v16h a0 = concat8(*(const v8h*)(kb + hi * 8),
                              *(const v8h*)(kb + 16 + hi * 8));
            v16h a1 = concat8(*(const v8h*)(kb + 32 + hi * 8),
                              *(const v8h*)(kb + 48 + hi * 8));
            v8f z = {};
            z = wmma_f16(a0, bq0, z);
            z = wmma_f16(a1, bq1, z);
            st[tt] = z;
        }
#endif
        // online softmax (lane holds query r; keys v+8*hi(+16) in VGPRs)
        float cmax = st[0][0];
#pragma unroll
        for (int v = 1; v < 8; ++v) cmax = fmaxf(cmax, st[0][v]);
#pragma unroll
        for (int v = 0; v < 8; ++v) cmax = fmaxf(cmax, st[1][v]);
        cmax = fmaxf(cmax, __shfl_xor(cmax, 16, 32));
        const float mnew  = fmaxf(m, cmax);
        const float alpha = exp2f((m - mnew) * cs);
        float csum = 0.f;
        v8f p0, p1;
#pragma unroll
        for (int v = 0; v < 8; ++v) { p0[v] = exp2f((st[0][v] - mnew) * cs); csum += p0[v]; }
#pragma unroll
        for (int v = 0; v < 8; ++v) { p1[v] = exp2f((st[1][v] - mnew) * cs); csum += p1[v]; }
        csum += __shfl_xor(csum, 16, 32);
        l = l * alpha + csum;
        m = mnew;

        // rescale ctx: ctx row = query v+8*hi -> broadcast alpha from lane v+8*hi
#pragma unroll
        for (int v = 0; v < 8; ++v) {
            float ab = __shfl(alpha, v + 8 * hi, 32);
            ctx[0][v] *= ab; ctx[1][v] *= ab; ctx[2][v] *= ab; ctx[3][v] *= ab;
        }
        // P in A-layout directly from scores^T C-layout
        v16h pa;
#pragma unroll
        for (int e = 0; e < 8; ++e) { pa[e] = (_Float16)p0[e]; pa[e + 8] = (_Float16)p1[e]; }

        // ctx += P(16x32) @ V(32x64)
#pragma unroll
        for (int nt = 0; nt < 4; ++nt) {
#if USE_ASYNC_LDS
            const _Float16* vp = &vsh[cur][(nt * 16 + r) * VPITCH + hi * 8];
            v16h bv = concat8(*(const v8h*)vp, *(const v8h*)(vp + 16));
#else
            v16h bv = concat8(*(const v8h*)(vbase0 + (size_t)(nt * 16 + r) * S_LEN + kc),
                              *(const v8h*)(vbase0 + (size_t)(nt * 16 + r) * S_LEN + kc + 16));
#endif
            ctx[nt] = wmma_f16(pa, bv, ctx[nt]);
        }
    }

    const float rl = 1.0f / l;
#pragma unroll
    for (int v = 0; v < 8; ++v) {
        float rb = __shfl(rl, v + 8 * hi, 32);
        int row = q0 + v + 8 * hi;
        _Float16* op = Ch + ((size_t)b * S_LEN + row) * D_MODEL + h * DK + r;
#pragma unroll
        for (int nt = 0; nt < 4; ++nt) op[nt * 16] = (_Float16)(ctx[nt][v] * rb);
    }
}

// -------------------------------------------------------------------------
extern "C" void kernel_launch(void* const* d_in, const int* in_sizes, int n_in,
                              void* d_out, int out_size, void* d_ws, size_t ws_size,
                              hipStream_t stream) {
    (void)in_sizes; (void)n_in; (void)out_size; (void)ws_size;
    const float* q    = (const float*)d_in[0];
    const float* k    = (const float*)d_in[1];
    const float* v    = (const float*)d_in[2];
    const float* Wq   = (const float*)d_in[3];
    const float* Wk   = (const float*)d_in[4];
    const float* Wv   = (const float*)d_in[5];
    const float* Wo   = (const float*)d_in[6];
    const float* gam  = (const float*)d_in[7];
    const float* bet  = (const float*)d_in[8];

    char* ws = (char*)d_ws;
    float*    Qf = (float*)ws;                             // 16 MB
    float*    Kf = (float*)(ws + (size_t)(16u << 20));     // 16 MB
    _Float16* Qh = (_Float16*)(ws + (size_t)(32u << 20));  //  8 MB
    _Float16* Kh = (_Float16*)(ws + (size_t)(40u << 20));  //  8 MB
    _Float16* Vt = (_Float16*)(ws + (size_t)(48u << 20));  //  8 MB
    _Float16* Xh = (_Float16*)(ws + (size_t)(56u << 20));  //  8 MB (input f16 / ctx f16)
    _Float16* Wh = (_Float16*)(ws + (size_t)(64u << 20));  //  2 MB (weight f16)

    dim3 blk(256);
    const int gx  = (M_ROWS * D_MODEL) / (256 * 8);        // 2048 blocks (4M elems)
    const int gw  = (D_MODEL * D_MODEL) / (256 * 8);       // 512 blocks (1M elems)

    // Q projection
    cvt_f16_kernel<<<gx, blk, 0, stream>>>(q, Xh);
    cvt_f16_kernel<<<gw, blk, 0, stream>>>(Wq, Wh);
    gemm_f16_kernel<<<256, blk, 0, stream>>>(Xh, Wh, Qf, nullptr, 0);
    // K projection
    cvt_f16_kernel<<<gx, blk, 0, stream>>>(k, Xh);
    cvt_f16_kernel<<<gw, blk, 0, stream>>>(Wk, Wh);
    gemm_f16_kernel<<<256, blk, 0, stream>>>(Xh, Wh, Kf, nullptr, 0);
    // V projection -> transposed f16 Vt
    cvt_f16_kernel<<<gx, blk, 0, stream>>>(v, Xh);
    cvt_f16_kernel<<<gw, blk, 0, stream>>>(Wv, Wh);
    gemm_f16_kernel<<<256, blk, 0, stream>>>(Xh, Wh, nullptr, Vt, 1);
    // shared QK LayerNorm -> f16 head layout
    ln_to_f16_kernel<<<512, blk, 0, stream>>>(Qf, gam, bet, Qh);
    ln_to_f16_kernel<<<512, blk, 0, stream>>>(Kf, gam, bet, Kh);
    // attention -> ctx f16 (reuse Xh buffer)
    attn_kernel<<<512, blk, 0, stream>>>(Qh, Kh, Vt, Xh);
    // output projection
    cvt_f16_kernel<<<gw, blk, 0, stream>>>(Wo, Wh);
    gemm_f16_kernel<<<256, blk, 0, stream>>>(Xh, Wh, (float*)d_out, nullptr, 0);
}
